// SelfAttention_34737695490244
// MI455X (gfx1250) — matrix operand order repaired and checked by reading, not compiled, not run on hardware
//
#include <hip/hip_runtime.h>
#include <hip/hip_bf16.h>
#include <math.h>

// ---------------------------------------------------------------------------
// Self-attention B=4, T=2048, D=1024, H=16, HD=64 on gfx1250 (wave32).
// Phase 1: f32 -> bf16 conversion of X and Wq/Wk/Wv.
// Phase 2: QKV projections as bf16 WMMA GEMMs (32x64 tile / wave, bias fused),
//          V stored transposed [b,h,d,t].
// Phase 3: flash-attention, 8 waves/block covering 128 query rows; K/V tiles
//          double-buffered into LDS with global_load_async_to_lds_b128
//          (ASYNCcnt) when available; WMMA for scores and P*V.
// Workspace: ~70 MB.
// ---------------------------------------------------------------------------

#define Bsz 4
#define Tsz 2048
#define Dsz 1024
#define Hsz 16
#define HDsz 64

typedef __attribute__((ext_vector_type(16))) __bf16 v16bf;
typedef __attribute__((ext_vector_type(8)))  __bf16 v8bf;
typedef __attribute__((ext_vector_type(8)))  float  v8f;
typedef __attribute__((ext_vector_type(4)))  int    v4i;

typedef __attribute__((address_space(1))) v4i  gas_v4i;   // global (AS1)
typedef __attribute__((address_space(3))) v4i  las_v4i;   // LDS (AS3)
typedef __attribute__((address_space(1))) void gas_void;
typedef __attribute__((address_space(3))) void las_void;

union V16U { v16bf v; v8bf h[2]; };

#if __has_builtin(__builtin_amdgcn_global_load_async_to_lds_b128)
#define HAVE_ASYNC_LDS 1
#else
#define HAVE_ASYNC_LDS 0
#endif

static __device__ __forceinline__ unsigned short f32_to_bf16(float f) {
  unsigned u = __float_as_uint(f);
  u += 0x7FFFu + ((u >> 16) & 1u);   // round-to-nearest-even
  return (unsigned short)(u >> 16);
}

// B-operand (K=32 x N=16, 16-bit): lane&15 = column, lane>=16 -> K offset +16,
// each lane holds 16 contiguous K elements (two 16-byte loads).
static __device__ __forceinline__ v16bf load_b_op(const unsigned short* p) {
  V16U r;
  r.h[0] = *(const v8bf*)(p);
  r.h[1] = *(const v8bf*)(p + 8);
  return r.v;
}

static __device__ __forceinline__ v8f wmma_bf16(v16bf a, v16bf b, v8f c) {
  return __builtin_amdgcn_wmma_f32_16x16x32_bf16(false, a, false, b,
                                                 (short)0, c, false, false);
}

// Stage 16 bytes global -> LDS (async when the builtin exists).
static __device__ __forceinline__ void stage16B(unsigned short* dst,
                                                const unsigned short* src) {
#if HAVE_ASYNC_LDS
  __builtin_amdgcn_global_load_async_to_lds_b128(
      (gas_v4i*)(gas_void*)(void*)const_cast<unsigned short*>(src),
      (las_v4i*)(las_void*)(void*)dst, 0, 0);
#else
  *(v8bf*)dst = *(const v8bf*)src;
#endif
}

static __device__ __forceinline__ void wait_stage() {
#if HAVE_ASYNC_LDS
#if __has_builtin(__builtin_amdgcn_s_wait_asynccnt)
  __builtin_amdgcn_s_wait_asynccnt(0);
#else
  asm volatile("s_wait_asynccnt 0x0" ::: "memory");
#endif
#endif
}

// ------------------------------ Phase 1 ------------------------------------
__global__ void cvt_f32_bf16(const float* __restrict__ src,
                             unsigned short* __restrict__ dst, int n) {
  int i = blockIdx.x * blockDim.x + threadIdx.x;
  if (i < n) dst[i] = f32_to_bf16(src[i]);
}

// ------------------------------ Phase 2 ------------------------------------
// OUT[m, j] = sum_k X[m,k]*W[j,k] + bias[j]   (torch Linear: x @ W.T + b)
// One wave computes a 32x64 output tile (2 row sub-tiles x 4 col sub-tiles).
// transpose_v == 0 : store [b, h, t, d] ; == 1 : store [b, h, d, t].
__global__ void qkv_gemm(const unsigned short* __restrict__ X,
                         const unsigned short* __restrict__ W,
                         const float* __restrict__ bias,
                         unsigned short* __restrict__ out,
                         int transpose_v) {
  const int lane = threadIdx.x;
  const int g  = lane >> 4;
  const int ln = lane & 15;
  const int m0 = blockIdx.x * 32;
  const int j0 = blockIdx.y * 64;

  v8f acc[2][4] = {};
  const unsigned short* xrow0 = X + (size_t)(m0 + ln) * Dsz;
  const unsigned short* xrow1 = X + (size_t)(m0 + 16 + ln) * Dsz;

  for (int k = 0; k < Dsz; k += 32) {
    V16U a0, a1;
    a0.h[0] = *(const v8bf*)(xrow0 + k + g * 16);
    a0.h[1] = *(const v8bf*)(xrow0 + k + g * 16 + 8);
    a1.h[0] = *(const v8bf*)(xrow1 + k + g * 16);
    a1.h[1] = *(const v8bf*)(xrow1 + k + g * 16 + 8);
#pragma unroll
    for (int i = 0; i < 4; ++i) {
      const v16bf b = load_b_op(W + (size_t)(j0 + 16 * i + ln) * Dsz + k + g * 16);
      acc[0][i] = wmma_bf16(a0.v, b, acc[0][i]);
      acc[1][i] = wmma_bf16(a1.v, b, acc[1][i]);
    }
  }

#pragma unroll
  for (int sub = 0; sub < 2; ++sub) {
#pragma unroll
    for (int i = 0; i < 4; ++i) {
      const int j = j0 + 16 * i + ln;
      const float bv = bias[j];
      const int h = j >> 6, d = j & 63;
#pragma unroll
      for (int r = 0; r < 8; ++r) {
        const int mrow = m0 + sub * 16 + r + g * 8;
        const int b = mrow >> 11, t = mrow & (Tsz - 1);
        const unsigned short val = f32_to_bf16(acc[sub][i][r] + bv);
        if (!transpose_v)
          out[((size_t)(b * Hsz + h) * Tsz + t) * HDsz + d] = val;
        else
          out[((size_t)(b * Hsz + h) * HDsz + d) * Tsz + t] = val;
      }
    }
  }
}

// ------------------------------ Phase 3 ------------------------------------
// grid: (T/128, H, B); block = 256 (8 waves, each owns 16 query rows).
// K/V tiles (32 keys) staged into LDS, double-buffered, async when possible.
__global__ void attn_fwd(const unsigned short* __restrict__ Q,   // [B,H,T,HD]
                         const unsigned short* __restrict__ K,   // [B,H,T,HD]
                         const unsigned short* __restrict__ VT,  // [B,H,HD,T]
                         const float* __restrict__ amask,        // [B,1,T,T]
                         const float* __restrict__ kpm,          // [B,T]
                         float* __restrict__ out) {              // [B,T,D]
  __shared__ unsigned short kbuf[2][32 * HDsz];   // [key][d], 4 KB each
  __shared__ unsigned short vbuf[2][HDsz * 32];   // [d][key], 4 KB each
  __shared__ unsigned short pbuf[8][16 * 32];     // per-wave P transpose tile

  const int tid  = threadIdx.x;
  const int wave = tid >> 5;
  const int lane = tid & 31;
  const int g  = lane >> 4;
  const int ln = lane & 15;
  const int t0 = blockIdx.x * 128 + wave * 16;
  const int h  = blockIdx.y;
  const int b  = blockIdx.z;
  const int bh = b * Hsz + h;

  const unsigned short* Kbh  = K  + (size_t)bh * Tsz * HDsz;
  const unsigned short* VTbh = VT + (size_t)bh * HDsz * Tsz;

  // Per-thread staging sources: 256 threads x 16B = one 4 KB tile each.
  const int vd = tid >> 2, vc = (tid & 3) * 8;   // V tile coords

  // Q A-operands.
  const unsigned short* qrow = Q + ((size_t)bh * Tsz + t0 + ln) * HDsz;
  V16U aq0, aq1;
  aq0.h[0] = *(const v8bf*)(qrow + g * 16);
  aq0.h[1] = *(const v8bf*)(qrow + g * 16 + 8);
  aq1.h[0] = *(const v8bf*)(qrow + 32 + g * 16);
  aq1.h[1] = *(const v8bf*)(qrow + 32 + g * 16 + 8);

  v8f acc[4] = {};
  float mrun[8], lrun[8];
#pragma unroll
  for (int r = 0; r < 8; ++r) { mrun[r] = -3.0e38f; lrun[r] = 0.0f; }

  const float scale = 0.125f;  // 1/sqrt(64)
  const float* mbase = amask + (size_t)b * Tsz * Tsz;
  const float* kbase = kpm + (size_t)b * Tsz;

  // Prologue: stage tile s0 = 0 into buffer 0.
  stage16B(&kbuf[0][tid * 8], Kbh + tid * 8);
  stage16B(&vbuf[0][vd * 32 + vc], VTbh + (size_t)vd * Tsz + vc);

  int cur = 0;
  for (int s0 = 0; s0 < Tsz; s0 += 32, cur ^= 1) {
    wait_stage();
    __syncthreads();   // tile `cur` visible to all waves

    if (s0 + 32 < Tsz) {   // prefetch next tile into the other buffer
      const int nb = cur ^ 1, ns = s0 + 32;
      stage16B(&kbuf[nb][tid * 8], Kbh + (size_t)ns * HDsz + tid * 8);
      stage16B(&vbuf[nb][vd * 32 + vc], VTbh + (size_t)vd * Tsz + ns + vc);
    }

    // ---- scores: two 16-key sub-tiles, K-dim 64 = 2 x 32 (from LDS) ------
    v8f slo = {}, shi = {};
    {
      const unsigned short* klo = &kbuf[cur][(ln)      * HDsz];
      const unsigned short* khi = &kbuf[cur][(16 + ln) * HDsz];
      slo = wmma_bf16(aq0.v, load_b_op(klo + g * 16), slo);
      slo = wmma_bf16(aq1.v, load_b_op(klo + 32 + g * 16), slo);
      shi = wmma_bf16(aq0.v, load_b_op(khi + g * 16), shi);
      shi = wmma_bf16(aq1.v, load_b_op(khi + 32 + g * 16), shi);
    }

    // ---- scale + attention mask + key padding ----------------------------
    const float kplo = (1.0f - kbase[s0 + ln]) * -10000.0f;
    const float kphi = (1.0f - kbase[s0 + 16 + ln]) * -10000.0f;
    float svlo[8], svhi[8], tmax[8];
#pragma unroll
    for (int r = 0; r < 8; ++r) {
      const int t = t0 + r + g * 8;
      const float* mr = mbase + (size_t)t * Tsz + s0 + ln;
      const float a = slo[r] * scale + mr[0]  + kplo;
      const float c = shi[r] * scale + mr[16] + kphi;
      svlo[r] = a; svhi[r] = c;
      tmax[r] = fmaxf(a, c);
    }
#pragma unroll
    for (int off = 1; off < 16; off <<= 1)
#pragma unroll
      for (int r = 0; r < 8; ++r)
        tmax[r] = fmaxf(tmax[r], __shfl_xor(tmax[r], off));

    // ---- online softmax update -------------------------------------------
    float psum[8];
#pragma unroll
    for (int r = 0; r < 8; ++r) {
      const float mnew  = fmaxf(mrun[r], tmax[r]);
      const float alpha = __expf(mrun[r] - mnew);
      mrun[r] = mnew;
      const float plo = __expf(svlo[r] - mnew);
      const float phi = __expf(svhi[r] - mnew);
      svlo[r] = plo; svhi[r] = phi;
      psum[r] = plo + phi;
      lrun[r] *= alpha;
      acc[0][r] *= alpha; acc[1][r] *= alpha;
      acc[2][r] *= alpha; acc[3][r] *= alpha;
    }
#pragma unroll
    for (int off = 1; off < 16; off <<= 1)
#pragma unroll
      for (int r = 0; r < 8; ++r)
        psum[r] += __shfl_xor(psum[r], off);
#pragma unroll
    for (int r = 0; r < 8; ++r) lrun[r] += psum[r];

    // ---- transpose P (C layout -> A layout) via per-wave LDS tile --------
    __syncthreads();
#pragma unroll
    for (int r = 0; r < 8; ++r) {
      const int row = r + g * 8;
      pbuf[wave][row * 32 + ln]      = f32_to_bf16(svlo[r]);
      pbuf[wave][row * 32 + 16 + ln] = f32_to_bf16(svhi[r]);
    }
    __syncthreads();
    V16U ap;
    const unsigned short* prow = &pbuf[wave][ln * 32];
    ap.h[0] = *(const v8bf*)(prow + g * 16);
    ap.h[1] = *(const v8bf*)(prow + g * 16 + 8);

    // ---- P (16x32) x V (32x64) accumulate (V from LDS) -------------------
#pragma unroll
    for (int i = 0; i < 4; ++i) {
      const unsigned short* vrow = &vbuf[cur][(i * 16 + ln) * 32 + g * 16];
      acc[i] = wmma_bf16(ap.v, load_b_op(vrow), acc[i]);
    }
  }

  // ---- normalize and store [b, t, h*64 + d] ------------------------------
#pragma unroll
  for (int r = 0; r < 8; ++r) {
    const float rinv = 1.0f / lrun[r];
    const int t = t0 + r + g * 8;
    float* orow = out + ((size_t)b * Tsz + t) * Dsz + h * HDsz;
#pragma unroll
    for (int i = 0; i < 4; ++i)
      orow[i * 16 + ln] = acc[i][r] * rinv;
  }
}

// ---------------------------------------------------------------------------
extern "C" void kernel_launch(void* const* d_in, const int* in_sizes, int n_in,
                              void* d_out, int out_size, void* d_ws, size_t ws_size,
                              hipStream_t stream) {
  const float* hs    = (const float*)d_in[0];
  const float* amask = (const float*)d_in[1];
  const float* kpmask= (const float*)d_in[2];
  const float* Wq    = (const float*)d_in[3];
  const float* bq    = (const float*)d_in[4];
  const float* Wk    = (const float*)d_in[5];
  const float* bk    = (const float*)d_in[6];
  const float* Wv    = (const float*)d_in[7];
  const float* bv    = (const float*)d_in[8];
  float* out = (float*)d_out;

  char* ws = (char*)d_ws;
  const size_t szX = (size_t)Bsz * Tsz * Dsz * 2;          // 16 MB
  const size_t szW = (size_t)Dsz * Dsz * 2;                //  2 MB
  const size_t szQ = (size_t)Bsz * Hsz * Tsz * HDsz * 2;   // 16 MB
  unsigned short* Xbf = (unsigned short*)(ws);
  unsigned short* Wqb = (unsigned short*)(ws + szX);
  unsigned short* Wkb = (unsigned short*)(ws + szX + szW);
  unsigned short* Wvb = (unsigned short*)(ws + szX + 2 * szW);
  unsigned short* Qb  = (unsigned short*)(ws + szX + 3 * szW);
  unsigned short* Kb  = (unsigned short*)(ws + szX + 3 * szW + szQ);
  unsigned short* VTb = (unsigned short*)(ws + szX + 3 * szW + 2 * szQ);

  {
    const int nX = Bsz * Tsz * Dsz;
    const int nW = Dsz * Dsz;
    cvt_f32_bf16<<<(nX + 255) / 256, 256, 0, stream>>>(hs, Xbf, nX);
    cvt_f32_bf16<<<(nW + 255) / 256, 256, 0, stream>>>(Wq, Wqb, nW);
    cvt_f32_bf16<<<(nW + 255) / 256, 256, 0, stream>>>(Wk, Wkb, nW);
    cvt_f32_bf16<<<(nW + 255) / 256, 256, 0, stream>>>(Wv, Wvb, nW);
  }
  {
    dim3 grid(Bsz * Tsz / 32, Dsz / 64);
    qkv_gemm<<<grid, 32, 0, stream>>>(Xbf, Wqb, bq, Qb, 0);
    qkv_gemm<<<grid, 32, 0, stream>>>(Xbf, Wkb, bk, Kb, 0);
    qkv_gemm<<<grid, 32, 0, stream>>>(Xbf, Wvb, bv, VTb, 1);
  }
  {
    dim3 grid(Tsz / 128, Hsz, Bsz);
    attn_fwd<<<grid, 256, 0, stream>>>(Qb, Kb, VTb, amask, kpmask, out);
  }
}
